// TTransformer_40432822124870
// MI455X (gfx1250) — compile-verified, matrix operand order checked by reading, not run on hardware
//
#include <hip/hip_runtime.h>
#include <hip/hip_bf16.h>

#define TN  288
#define EMB 512
#define NH  8
#define HD  64
#define NB  64
#define TT  512

#ifndef __has_builtin
#define __has_builtin(x) 0
#endif
#if __has_builtin(__builtin_amdgcn_tensor_load_to_lds) && \
    __has_builtin(__builtin_amdgcn_s_wait_tensorcnt)
#define HAVE_TDM 1
#else
#define HAVE_TDM 0
#endif

typedef __attribute__((ext_vector_type(16))) __bf16 v16bf;
typedef __attribute__((ext_vector_type(8)))  float  v8f;
typedef __attribute__((ext_vector_type(4)))  unsigned int u32x4;
typedef __attribute__((ext_vector_type(4)))  unsigned int v4u;
typedef __attribute__((ext_vector_type(8)))  int v8i;
typedef __attribute__((ext_vector_type(4)))  int v4i;

union FragBF { v16bf v; __bf16 e[16]; unsigned u[8]; u32x4 q2[2]; };

__device__ __forceinline__ __bf16 f2bf(float f) {
  unsigned u = __builtin_bit_cast(unsigned, f);
  unsigned r = u + 0x7FFFu + ((u >> 16) & 1u);
  unsigned short h = (unsigned short)(r >> 16);
  return __builtin_bit_cast(__bf16, h);
}
// pack two f32 -> two bf16 in one dword (lo, hi)
__device__ __forceinline__ unsigned f2bf2(float lo, float hi) {
  unsigned ulo = __builtin_bit_cast(unsigned, lo);
  unsigned uhi = __builtin_bit_cast(unsigned, hi);
  ulo += 0x7FFFu + ((ulo >> 16) & 1u);
  uhi += 0x7FFFu + ((uhi >> 16) & 1u);
  return (ulo >> 16) | (uhi & 0xFFFF0000u);
}
__device__ __forceinline__ float bf2f(__bf16 b) {
  unsigned short h = __builtin_bit_cast(unsigned short, b);
  unsigned u = ((unsigned)h) << 16;
  return __builtin_bit_cast(float, u);
}

__device__ __forceinline__ v8f wmma_bf16(v16bf a, v16bf b, v8f c) {
  return __builtin_amdgcn_wmma_f32_16x16x32_bf16(false, a, false, b, (short)0, c,
                                                 false, false);
}

// A-frag: 16 rows x 32 k from row-major src (row stride ld elems).
__device__ __forceinline__ v16bf load_a_frag(const __bf16* src, int ld) {
  int lane = threadIdx.x & 31;
  int row = lane & 15, kh = lane >> 4;
  const __bf16* p = src + row * ld + kh * 8;
  FragBF f;
  f.q2[0] = *(const u32x4*)p;
  f.q2[1] = *(const u32x4*)(p + 16);
  return f.v;
}

// B-frag: 32k x 16n where memory Mt is n-major: Mt[n*ldk + k]
__device__ __forceinline__ v16bf load_b_frag_t(const __bf16* Mt, int ldk) {
  int lane = threadIdx.x & 31;
  int col = lane & 15, kh = lane >> 4;
  const __bf16* p = Mt + col * ldk + kh * 8;
  FragBF f;
  f.q2[0] = *(const u32x4*)p;
  f.q2[1] = *(const u32x4*)(p + 16);
  return f.v;
}

#if HAVE_TDM
// 2-D TDM load: dim0 contiguous elems (bf16), dim1 rows, row stride in elems.
__device__ __forceinline__ void tdm_load_2d(unsigned lds_off, const void* gaddr,
                                            unsigned dim0, unsigned dim1,
                                            unsigned stride0) {
  unsigned long long ga = (unsigned long long)gaddr;
  v4u g0;
  g0[0] = 1u;                                           // count=1
  g0[1] = lds_off;                                      // lds_addr (bytes)
  g0[2] = (unsigned)(ga & 0xFFFFFFFFu);                 // global_addr lo
  g0[3] = (unsigned)((ga >> 32) & 0x01FFFFFFu) | (2u << 30);  // hi + type=2
  v8i g1;
  g1[0] = (int)(1u << 16);                              // data_size=2B
  g1[1] = (int)(dim0 << 16);                            // tensor_dim0 lo16
  g1[2] = (int)((dim0 >> 16) | ((dim1 & 0xFFFFu) << 16));
  g1[3] = (int)((dim1 >> 16) | (dim0 << 16));           // + tile_dim0
  g1[4] = (int)(dim1 & 0xFFFFu);                        // tile_dim1
  g1[5] = (int)stride0;                                 // dim0 stride lo32
  g1[6] = 0;
  g1[7] = 0;
  v4i z4 = {0, 0, 0, 0};
#if __clang_major__ >= 23
  v8i z8 = {0, 0, 0, 0, 0, 0, 0, 0};
  __builtin_amdgcn_tensor_load_to_lds(g0, g1, z4, z4, z8, 0);
#else
  __builtin_amdgcn_tensor_load_to_lds(g0, g1, z4, z4, 0);
#endif
}
#endif

// ---------------- Kernel 1: positional one-hot encode --------------------
__global__ void pos_onehot_kernel(const float* __restrict__ I_mat,
                                  const float* __restrict__ ohW,
                                  const float* __restrict__ ohb,
                                  const int* __restrict__ ip,
                                  float* __restrict__ onehot) {
  int t = blockIdx.x;
  int e = threadIdx.x;
  int row = (ip[0] + t) % TN;
  float acc = ohb[e];
  for (int k = 0; k < TN; ++k)
    acc = fmaf(I_mat[row * TN + k], ohW[k * EMB + e], acc);
  onehot[t * EMB + e] = acc;
}

// ---------------- Kernel 2: q_in = query + onehot (bf16) -----------------
__global__ void add_pos_kernel(const float* __restrict__ query,
                               const float* __restrict__ onehot,
                               __bf16* __restrict__ q_in) {
  long idx = (long)blockIdx.x * blockDim.x + threadIdx.x;
  int te = (int)(idx % ((long)TT * EMB));
  q_in[idx] = f2bf(query[idx] + onehot[te]);
}

// ---------------- Kernel 3: head projection GEMM (M x 64) @ (64 x 64) ----
// Weight staged once per block in LDS (n-major); B-frags loaded transiently.
__global__ __launch_bounds__(256, 1) void proj_gemm_kernel(
    const float* __restrict__ Af, const __bf16* __restrict__ Ab,
    const float* __restrict__ W, __bf16* __restrict__ out, float scale) {
  __shared__ __bf16 WT[64 * 64];  // WT[n*64 + k] = W[k*64 + n]
  for (int idx = threadIdx.x; idx < 64 * 64; idx += 256) {
    int k = idx >> 6, n = idx & 63;
    WT[n * 64 + k] = f2bf(W[idx]);
  }
  __syncthreads();

  int wave = threadIdx.x >> 5;
  long r0 = ((long)blockIdx.x * 8 + wave) * 16;
  int lane = threadIdx.x & 31;
  int col = lane & 15, kh = lane >> 4;
  int row = lane & 15;

  v16bf afr[2];
#pragma unroll
  for (int kb2 = 0; kb2 < 2; ++kb2) {
    FragBF f;
    int k0 = kb2 * 32 + kh * 8;
    if (Af) {
      const float* p = Af + (r0 + row) * 64 + k0;
#pragma unroll
      for (int i = 0; i < 4; ++i)
        f.u[i] = f2bf2(p[2 * i], p[2 * i + 1]);
#pragma unroll
      for (int i = 0; i < 4; ++i)
        f.u[4 + i] = f2bf2(p[16 + 2 * i], p[16 + 2 * i + 1]);
    } else {
      const __bf16* p = Ab + (r0 + row) * 64 + k0;
      f.q2[0] = *(const u32x4*)p;
      f.q2[1] = *(const u32x4*)(p + 16);
    }
    afr[kb2] = f.v;
  }

#pragma unroll
  for (int nb = 0; nb < 4; ++nb) {
    v8f acc = {};
    acc = wmma_bf16(afr[0], load_b_frag_t(WT + nb * 16 * 64, 64), acc);
    acc = wmma_bf16(afr[1], load_b_frag_t(WT + nb * 16 * 64 + 32, 64), acc);
    int rbase = kh * 8;
#pragma unroll
    for (int r = 0; r < 8; ++r)
      out[(r0 + rbase + r) * 64 + nb * 16 + col] = f2bf(acc[r] * scale);
  }
}

// ---------------- Kernel 4: flash attention per (n,h,qtile=128) ----------
__global__ __launch_bounds__(256, 1) void attention_kernel(
    const __bf16* __restrict__ qh, const __bf16* __restrict__ kb,
    const __bf16* __restrict__ vb, __bf16* __restrict__ attn) {
  extern __shared__ char smem[];
  __bf16* Kl = (__bf16*)smem;            // 512 x 64  (LDS offset 0)
  __bf16* VT = Kl + TT * HD;             // 64 x 512 (d-major)
  __bf16* Pl = VT + HD * TT;             // 8 waves x (16 x 32)

  int qt = blockIdx.x & 3;
  int nh = blockIdx.x >> 2;
  int n = nh >> 3, h = nh & 7;

  const __bf16* Kg = kb + ((long)n * TT * NH + h) * HD;
  const __bf16* Vg = vb + ((long)n * TT * NH + h) * HD;

#if HAVE_TDM
  // TDM: async K tile (512 rows x 64 bf16, row stride EMB) -> Kl
  if ((threadIdx.x >> 5) == 0) tdm_load_2d(0u, Kg, HD, TT, EMB);
#endif
  // cooperative V transpose into LDS (and K copy if no TDM)
  for (int idx4 = threadIdx.x * 4; idx4 < TT * HD; idx4 += 256 * 4) {
    int t0 = idx4 >> 6, d0 = idx4 & 63;
#if !HAVE_TDM
    *(unsigned long long*)(Kl + t0 * 64 + d0) =
        *(const unsigned long long*)(Kg + (long)t0 * EMB + d0);
#endif
    __bf16 vv[4];
    *(unsigned long long*)vv =
        *(const unsigned long long*)(Vg + (long)t0 * EMB + d0);
#pragma unroll
    for (int j = 0; j < 4; ++j) VT[(d0 + j) * TT + t0] = vv[j];
  }
#if HAVE_TDM
  if ((threadIdx.x >> 5) == 0) __builtin_amdgcn_s_wait_tensorcnt(0);
#endif
  __syncthreads();

  int wave = threadIdx.x >> 5;
  int lane = threadIdx.x & 31;
  int col = lane & 15, rb = (lane >> 4) * 8;
  int q0 = qt * 128 + wave * 16;
  __bf16* Pw = Pl + wave * 16 * 32;

  const __bf16* Qg = qh + ((long)(n * TT + q0) * NH + h) * HD;
  v16bf aq0 = load_a_frag(Qg, EMB);       // d 0..31
  v16bf aq1 = load_a_frag(Qg + 32, EMB);  // d 32..63

  float Mrow[8], Lrow[8];
  v8f o[4];
#pragma unroll
  for (int r = 0; r < 8; ++r) { Mrow[r] = -1e30f; Lrow[r] = 0.f; }
#pragma unroll
  for (int nb = 0; nb < 4; ++nb) { v8f z = {}; o[nb] = z; }

  for (int kb0 = 0; kb0 < TT / 32; ++kb0) {
    const __bf16* Kp = Kl + kb0 * 32 * HD;
    v8f s0 = {}, s1 = {};
    s0 = wmma_bf16(aq0, load_b_frag_t(Kp, HD), s0);
    s0 = wmma_bf16(aq1, load_b_frag_t(Kp + 32, HD), s0);
    s1 = wmma_bf16(aq0, load_b_frag_t(Kp + 16 * HD, HD), s1);
    s1 = wmma_bf16(aq1, load_b_frag_t(Kp + 16 * HD + 32, HD), s1);

    float bm[8], p0e[8], p1e[8], ps[8];
#pragma unroll
    for (int r = 0; r < 8; ++r) bm[r] = fmaxf(s0[r], s1[r]);
#pragma unroll
    for (int off = 1; off < 16; off <<= 1)
#pragma unroll
      for (int r = 0; r < 8; ++r)
        bm[r] = fmaxf(bm[r], __shfl_xor(bm[r], off, 32));
#pragma unroll
    for (int r = 0; r < 8; ++r) {
      float nm = fmaxf(Mrow[r], bm[r]);
      float al = __expf(Mrow[r] - nm);
      Mrow[r] = nm;
      Lrow[r] *= al;
#pragma unroll
      for (int nb = 0; nb < 4; ++nb) o[nb][r] = o[nb][r] * al;
      p0e[r] = __expf(s0[r] - nm);
      p1e[r] = __expf(s1[r] - nm);
      ps[r] = p0e[r] + p1e[r];
    }
#pragma unroll
    for (int off = 1; off < 16; off <<= 1)
#pragma unroll
      for (int r = 0; r < 8; ++r) ps[r] += __shfl_xor(ps[r], off, 32);
#pragma unroll
    for (int r = 0; r < 8; ++r) Lrow[r] += ps[r];

    // C-layout -> A-layout via per-wave LDS round trip
#pragma unroll
    for (int r = 0; r < 8; ++r) {
      Pw[(rb + r) * 32 + col]      = f2bf(p0e[r]);
      Pw[(rb + r) * 32 + 16 + col] = f2bf(p1e[r]);
    }
    v16bf ap = load_a_frag(Pw, 32);
#pragma unroll
    for (int nb = 0; nb < 4; ++nb)
      o[nb] = wmma_bf16(ap, load_b_frag_t(VT + (nb * 16) * TT + kb0 * 32, TT),
                        o[nb]);
  }

#pragma unroll
  for (int r = 0; r < 8; ++r) {
    float inv = 1.0f / Lrow[r];
    long base = ((long)(n * TT + q0 + rb + r) * NH + h) * HD;
#pragma unroll
    for (int nb = 0; nb < 4; ++nb)
      attn[base + nb * 16 + col] = f2bf(o[nb][r] * inv);
  }
}

// ---------------- Kernel 5: tiled WMMA GEMM with fused epilogue ----------
template <bool RELU, bool OUTBF>
__global__ __launch_bounds__(256, 1) void gemm_kernel(
    const __bf16* __restrict__ A, const float* __restrict__ Bsrc,
    const float* __restrict__ bias, const __bf16* __restrict__ residb,
    const float* __restrict__ residf, float* __restrict__ outf,
    __bf16* __restrict__ outb, int K, int Ntot) {
  extern __shared__ char smem[];
  __bf16* As   = (__bf16*)smem;       // 32 x 32
  __bf16* BsmT = As + 32 * 32;        // 512 x 32 (n-major)

  int mbase = blockIdx.x * 32;
  int nbase = blockIdx.y * 512;
  int wave = threadIdx.x >> 5;
  int wr = wave >> 2, wc = wave & 3;
  int lane = threadIdx.x & 31;
  int col = lane & 15, rb = (lane >> 4) * 8;

  v8f acc[8];
#pragma unroll
  for (int c = 0; c < 8; ++c) { v8f z = {}; acc[c] = z; }

  for (int kk = 0; kk < K; kk += 32) {
    // stage A tile (32 x 32 bf16)
    {
      int idx4 = threadIdx.x * 4;
      int ar = idx4 >> 5, ac = idx4 & 31;
      *(unsigned long long*)(As + ar * 32 + ac) =
          *(const unsigned long long*)(A + (long)(mbase + ar) * K + kk + ac);
    }
    if (kk + 32 < K)
      __builtin_prefetch(Bsrc + (long)(kk + 32) * Ntot + nbase, 0, 1);
    // stage B tile transposed (512 x 32), f32 -> bf16
#pragma unroll 4
    for (int it = 0; it < 16; ++it) {
      int idx4 = (threadIdx.x + it * 256) * 4;
      int krow = idx4 >> 9, ncol = idx4 & 511;
      const float4 b4 =
          *(const float4*)(Bsrc + (long)(kk + krow) * Ntot + nbase + ncol);
      BsmT[(ncol + 0) * 32 + krow] = f2bf(b4.x);
      BsmT[(ncol + 1) * 32 + krow] = f2bf(b4.y);
      BsmT[(ncol + 2) * 32 + krow] = f2bf(b4.z);
      BsmT[(ncol + 3) * 32 + krow] = f2bf(b4.w);
    }
    __syncthreads();

    v16bf af = load_a_frag(As + wr * 16 * 32, 32);
#pragma unroll
    for (int c = 0; c < 8; ++c) {
      v16bf bf = load_b_frag_t(BsmT + (wc * 128 + c * 16) * 32, 32);
      acc[c] = wmma_bf16(af, bf, acc[c]);
    }
    __syncthreads();
  }

#pragma unroll
  for (int c = 0; c < 8; ++c) {
    int colabs = nbase + wc * 128 + c * 16 + col;
    float bv = bias[colabs];
#pragma unroll
    for (int r = 0; r < 8; ++r) {
      long row = mbase + wr * 16 + rb + r;
      float v = acc[c][r] + bv;
      if (residb) v += bf2f(residb[row * 512 + colabs]);
      if (residf) v += residf[row * 512 + colabs];
      if (RELU) v = fmaxf(v, 0.f);
      if (OUTBF) outb[row * (long)Ntot + colabs] = f2bf(v);
      else       outf[row * (long)Ntot + colabs] = v;
    }
  }
}

// ---------------- Kernel 6: per-row LayerNorm (one wave per row) ---------
__global__ void ln_kernel(const float* __restrict__ src,
                          const float* __restrict__ g,
                          const float* __restrict__ b,
                          float* __restrict__ dst,
                          __bf16* __restrict__ dstb) {
  int row = blockIdx.x * (blockDim.x >> 5) + (threadIdx.x >> 5);
  int lane = threadIdx.x & 31;
  const float* p = src + (long)row * EMB;
  float v[16], s = 0.f, s2 = 0.f;
#pragma unroll
  for (int i = 0; i < 16; ++i) {
    v[i] = p[lane + i * 32];
    s += v[i];
    s2 += v[i] * v[i];
  }
#pragma unroll
  for (int off = 1; off < 32; off <<= 1) {
    s += __shfl_xor(s, off, 32);
    s2 += __shfl_xor(s2, off, 32);
  }
  float mean = s * (1.0f / EMB);
  float var = s2 * (1.0f / EMB) - mean * mean;
  float rstd = rsqrtf(var + 1e-5f);
#pragma unroll
  for (int i = 0; i < 16; ++i) {
    int c = lane + i * 32;
    float o = (v[i] - mean) * rstd * g[c] + b[c];
    dst[(long)row * EMB + c] = o;
    if (dstb) dstb[(long)row * EMB + c] = f2bf(o);
  }
}

extern "C" void kernel_launch(void* const* d_in, const int* in_sizes, int n_in,
                              void* d_out, int out_size, void* d_ws,
                              size_t ws_size, hipStream_t stream) {
  const float* value  = (const float*)d_in[0];
  const float* key_in = (const float*)d_in[1];
  const float* query  = (const float*)d_in[2];
  const int*   ip     = (const int*)d_in[3];
  const float* I_mat  = (const float*)d_in[4];
  const float* ohW    = (const float*)d_in[5];
  const float* ohb    = (const float*)d_in[6];
  const float* Wv     = (const float*)d_in[7];
  const float* Wk     = (const float*)d_in[8];
  const float* Wq     = (const float*)d_in[9];
  const float* Wo     = (const float*)d_in[10];
  const float* bo     = (const float*)d_in[11];
  const float* g1     = (const float*)d_in[12];
  const float* b1     = (const float*)d_in[13];
  const float* g2     = (const float*)d_in[14];
  const float* b2     = (const float*)d_in[15];
  const float* W1     = (const float*)d_in[16];
  const float* bf1    = (const float*)d_in[17];
  const float* W2     = (const float*)d_in[18];
  const float* bf2    = (const float*)d_in[19];

  const long NTE = (long)NB * TT * EMB;   // 16,777,216
  char* ws = (char*)d_ws;
  float*  onehot = (float*)ws;                       // 1 MB
  __bf16* q_in   = (__bf16*)(ws + (1l << 20));       // 32 MB each bf16 buf
  __bf16* qh     = q_in + NTE;
  __bf16* kbuf   = qh + NTE;
  __bf16* vbuf   = kbuf + NTE;
  __bf16* attn   = vbuf + NTE;
  float*  tmp    = (float*)(attn + NTE);             // 64 MB f32
  __bf16* xb     = (__bf16*)(tmp + NTE);             // 32 MB
  __bf16* ffh    = xb + NTE;                         // 128 MB
  float*  out    = (float*)d_out;

  pos_onehot_kernel<<<TT, EMB, 0, stream>>>(I_mat, ohW, ohb, ip, onehot);
  add_pos_kernel<<<(int)(NTE / 256), 256, 0, stream>>>(query, onehot, q_in);

  // M = N*T*H = 262144 rows, 16 rows/wave, 8 waves/block -> 2048 blocks
  proj_gemm_kernel<<<2048, 256, 0, stream>>>(value, nullptr, Wv, vbuf, 1.0f);
  proj_gemm_kernel<<<2048, 256, 0, stream>>>(key_in, nullptr, Wk, kbuf, 1.0f);
  proj_gemm_kernel<<<2048, 256, 0, stream>>>(nullptr, q_in, Wq, qh,
                                             0.0441941738f /* 1/sqrt(512) */);

  size_t attn_lds = (size_t)(TT * HD + HD * TT + 8 * 16 * 32) * 2;
  attention_kernel<<<NB * NH * (TT / 128), 256, attn_lds, stream>>>(qh, kbuf,
                                                                    vbuf, attn);

  size_t gemm_lds = (size_t)(32 * 32 + 512 * 32) * 2;
  // out = attn @ Wo + bo + q_in  -> tmp (f32)
  gemm_kernel<false, false><<<dim3(1024, 1), 256, gemm_lds, stream>>>(
      attn, Wo, bo, q_in, nullptr, tmp, nullptr, 512, 512);
  // x = LN1(tmp) in-place; also bf16 copy xb
  ln_kernel<<<4096, 256, 0, stream>>>(tmp, g1, b1, tmp, xb);
  // ffh = relu(xb @ W1 + bf1) (bf16)
  gemm_kernel<true, true><<<dim3(1024, 4), 256, gemm_lds, stream>>>(
      xb, W1, bf1, nullptr, nullptr, nullptr, ffh, 512, 2048);
  // out = ffh @ W2 + bf2 + x  -> d_out (f32)
  gemm_kernel<false, false><<<dim3(1024, 1), 256, gemm_lds, stream>>>(
      ffh, W2, bf2, nullptr, tmp, out, nullptr, 2048, 512);
  // final = LN2(d_out) in-place
  ln_kernel<<<4096, 256, 0, stream>>>(out, g2, b2, out, nullptr);
}
